// BertSelfAttention_76390288327166
// MI455X (gfx1250) — matrix operand order repaired and checked by reading, not compiled, not run on hardware
//
#include <hip/hip_runtime.h>
#include <hip/hip_bf16.h>
#include <math.h>

// ---------------------------------------------------------------------------
// BERT-style 2D self-attention for MI455X (gfx1250), wave32 + WMMA f16 + TDM.
// ---------------------------------------------------------------------------

typedef _Float16 h16;
typedef __attribute__((ext_vector_type(16))) _Float16 v16h;
typedef __attribute__((ext_vector_type(8)))  _Float16 h8;
typedef __attribute__((ext_vector_type(8)))  float    v8f;
typedef __attribute__((ext_vector_type(4)))  float    f4;
typedef __attribute__((ext_vector_type(4)))  unsigned int v4u;
typedef __attribute__((ext_vector_type(8)))  int      v8i;
typedef __attribute__((ext_vector_type(4)))  int      v4i;

#define HEADS 8
#define MD    64
#define DIM   512
#define SEQ   1024
#define BATCH 8
#define ROWS  (BATCH * SEQ)     // 8192
#define EPSF  1e-5f

// workspace byte offsets
#define OFF_XN   (size_t)0                        // xn f32   : 16 MB
#define OFF_XNH  (size_t)16777216                 // xn f16   :  8 MB
#define OFF_QKV  (size_t)25165824                 // q,k,v f16: 24 MB (v stored transposed)
#define OFF_LOG  (size_t)50331648                 // logits f16: 8 MB
#define OFF_WH   (size_t)58720256                 // Wq^T,Wk^T,Wv^T,Wo^T f16: 2 MB

// Tensor Data Mover availability (this toolchain: 6-arg builtin,
// (v4u g0, v8i g1, v4i g2, v4i g3, v8i g4, i32 cpol))
#if defined(__has_builtin)
#if __has_builtin(__builtin_amdgcn_tensor_load_to_lds) && \
    __has_builtin(__builtin_amdgcn_s_wait_tensorcnt)
#define USE_TDM 1
#endif
#endif
#ifndef USE_TDM
#define USE_TDM 0
#endif

// ---------------------------------------------------------------------------
// Fragment loaders (ISA 7.12.2 layouts, wave32):
//   A 16x32 f16 : lane<16 -> elems0..7 = K0..7,  elems8..15 = K16..23
//                 lane>=16 -> elems0..7 = K8..15, elems8..15 = K24..31
//   B 32x16 f16 : same K order as A (N = lane&15)
//   C/D 16x16 f32: VGPR r -> M = r + 8*(lane>=16), N = lane&15
// B tiles are staged K-contiguous per column, so both loaders are two aligned
// 16-byte ds_load_b128.
// ---------------------------------------------------------------------------
__device__ __forceinline__ v16h load_a_frag(const h16* smem, int row, int stride,
                                            int kk, int hs) {
  const h16* p = smem + row * stride + kk;
  h8 lo = *(const h8*)(p + hs * 8);
  h8 hi = *(const h8*)(p + 16 + hs * 8);
  v16h a;
#pragma unroll
  for (int i = 0; i < 8; ++i) { a[i] = lo[i]; a[i + 8] = hi[i]; }
  return a;
}

__device__ __forceinline__ v16h load_b_frag(const h16* smemT, int col, int stride,
                                            int kk, int hs) {
  const h16* p = smemT + col * stride + kk;
  h8 lo = *(const h8*)(p + hs * 8);
  h8 hi = *(const h8*)(p + 16 + hs * 8);
  v16h b;
#pragma unroll
  for (int i = 0; i < 8; ++i) { b[i] = lo[i]; b[i + 8] = hi[i]; }
  return b;
}

#define WMMA_F16(A, B, C) \
  __builtin_amdgcn_wmma_f32_16x16x32_f16(false, (A), false, (B), (short)0, (C), false, false)

// ---------------------------------------------------------------------------
// 1) LayerNorm: one block per row of 512; writes f32 (residual) + f16 (GEMM).
// ---------------------------------------------------------------------------
__global__ __launch_bounds__(256) void ln_kernel(const float* __restrict__ x,
                                                 const float* __restrict__ gamma,
                                                 const float* __restrict__ beta,
                                                 float* __restrict__ xn,
                                                 h16* __restrict__ xnh) {
  __shared__ float red[256];
  const int row = blockIdx.x;
  const int t = threadIdx.x;
  const float* xr = x + (size_t)row * DIM;
  float a = xr[t], b = xr[t + 256];
  red[t] = a + b;
  __syncthreads();
#pragma unroll
  for (int off = 128; off > 0; off >>= 1) {
    if (t < off) red[t] += red[t + off];
    __syncthreads();
  }
  const float mu = red[0] * (1.0f / DIM);
  __syncthreads();
  const float da = a - mu, db = b - mu;
  red[t] = da * da + db * db;
  __syncthreads();
#pragma unroll
  for (int off = 128; off > 0; off >>= 1) {
    if (t < off) red[t] += red[t + off];
    __syncthreads();
  }
  const float rstd = rsqrtf(red[0] * (1.0f / DIM) + EPSF);
  const float y0 = da * rstd * gamma[t] + beta[t];
  const float y1 = db * rstd * gamma[t + 256] + beta[t + 256];
  float* xo = xn + (size_t)row * DIM;
  h16* ho = xnh + (size_t)row * DIM;
  xo[t] = y0;        xo[t + 256] = y1;
  ho[t] = (h16)y0;   ho[t + 256] = (h16)y1;
}

// ---------------------------------------------------------------------------
// 2) Weight cast + transpose: wh[z][n][k] = (f16) W_z[k][n].
//    One-time 2 MB; makes all GEMM B-tiles K-contiguous in global memory.
// ---------------------------------------------------------------------------
__global__ __launch_bounds__(256) void wcast_kernel(const float* __restrict__ Wq,
                                                    const float* __restrict__ Wk,
                                                    const float* __restrict__ Wv,
                                                    const float* __restrict__ Wo,
                                                    h16* __restrict__ wh) {
  const int i = blockIdx.x * 256 + threadIdx.x;        // < 4*262144
  const int z = i >> 18;
  const int r = i & 262143;
  const int n = r >> 9, k = r & 511;
  const float* s = (z == 0) ? Wq : (z == 1) ? Wk : (z == 2) ? Wv : Wo;
  wh[i] = (h16)s[(size_t)k * DIM + n];
}

// ---------------------------------------------------------------------------
// 3) Q/K/V projection: xn_f16[8192x512] @ W[512x512] -> f16.
//    Block tile 128x64, BK=32; 8 waves (4 M x 2 N), wave = 32x32 = 2x2 WMMA.
//    z selects Wq/Wk/Wv.  z==2 (V) is stored TRANSPOSED per (b,h):
//      vT[(b*512 + h*64 + d)*1024 + token]   (epilogue is scattered anyway).
// ---------------------------------------------------------------------------
__global__ __launch_bounds__(256) void gemm_qkv_kernel(const h16* __restrict__ xnh,
                                                       const h16* __restrict__ wh,
                                                       h16* __restrict__ qkv) {
  __shared__ __attribute__((aligned(16))) h16 As[128 * 32];
  __shared__ __attribute__((aligned(16))) h16 Bt[64 * 32];
  const int mBase = blockIdx.x * 128;
  const int nBase = blockIdx.y * 64;
  const int z = blockIdx.z;
  const h16* __restrict__ Wt = wh + (size_t)z * DIM * DIM;   // [n][k]
  h16* __restrict__ out = qkv + (size_t)z * ROWS * DIM;
  const int t = threadIdx.x;
  const int lane = t & 31, wv = t >> 5;
  const int wm = (wv & 3) * 32, wn = (wv >> 2) * 32;
  const int hs = lane >> 4, lr = lane & 15;

  v8f acc[2][2] = {};
  for (int kk = 0; kk < DIM; kk += 32) {
    {  // A tile 128x32 : 2 threads/row, 16 halfs each
      const int r = t >> 1, c = (t & 1) * 16;
      const h16* g = xnh + (size_t)(mBase + r) * DIM + kk + c;
      *(h8*)&As[r * 32 + c] = *(const h8*)(g);
      *(h8*)&As[r * 32 + c + 8] = *(const h8*)(g + 8);
      if (kk + 32 < DIM) __builtin_prefetch(g + 32, 0, 3);   // global_prefetch_b8
    }
    {  // B tile: already transposed in global -> contiguous b128 stage
      const int n = t >> 2, ko = (t & 3) * 8;
      *(h8*)&Bt[n * 32 + ko] = *(const h8*)(Wt + (size_t)(nBase + n) * DIM + kk + ko);
    }
    __syncthreads();
    v16h af[2], bf[2];
#pragma unroll
    for (int mi = 0; mi < 2; ++mi) af[mi] = load_a_frag(As, wm + mi * 16 + lr, 32, 0, hs);
#pragma unroll
    for (int ni = 0; ni < 2; ++ni) bf[ni] = load_b_frag(Bt, wn + ni * 16 + lr, 32, 0, hs);
#pragma unroll
    for (int mi = 0; mi < 2; ++mi)
#pragma unroll
      for (int ni = 0; ni < 2; ++ni) acc[mi][ni] = WMMA_F16(af[mi], bf[ni], acc[mi][ni]);
    __syncthreads();
  }
#pragma unroll
  for (int mi = 0; mi < 2; ++mi)
#pragma unroll
    for (int ni = 0; ni < 2; ++ni) {
      const int col = nBase + wn + ni * 16 + lr;
      const int rB = mBase + wm + mi * 16 + hs * 8;
      if (z == 2) {  // V: transposed layout for av_kernel staging
#pragma unroll
        for (int r = 0; r < 8; ++r) {
          const int rG = rB + r;
          out[((size_t)(rG >> 10) * 512 + col) * SEQ + (rG & 1023)] = (h16)acc[mi][ni][r];
        }
      } else {
#pragma unroll
        for (int r = 0; r < 8; ++r)
          out[(size_t)(rB + r) * DIM + col] = (h16)acc[mi][ni][r];
      }
    }
}

// ---------------------------------------------------------------------------
// 4) Scores: per (b,h) Q·K^T / 8 -> f32 into d_out's attn region.
//    Block tile 128(q) x 128(k), K-dim=64.  Q/K tiles fetched with the Tensor
//    Data Mover when available (2D D#: tile 64x128 halfs, row stride 512),
//    one DMA from wave0 (Q) and wave1 (K); else manual b128 staging.
// ---------------------------------------------------------------------------
__global__ __launch_bounds__(256) void scores_kernel(const h16* __restrict__ q,
                                                     const h16* __restrict__ k,
                                                     float* __restrict__ corr) {
  __shared__ __attribute__((aligned(16))) h16 Qs[128 * 64];
  __shared__ __attribute__((aligned(16))) h16 Ks[128 * 64];
  const int qBase = blockIdx.x * 128;
  const int kBase = blockIdx.y * 128;
  const int bh = blockIdx.z;
  const int b = bh >> 3, h = bh & 7;
  const int t = threadIdx.x;
  const int lane = t & 31, wv = t >> 5;

#if USE_TDM
  if (wv < 2) {
    const h16* gbase = (wv == 0)
        ? q + (size_t)(b * SEQ + qBase) * DIM + h * MD
        : k + (size_t)(b * SEQ + kBase) * DIM + h * MD;
    const unsigned long long ga = (unsigned long long)(uintptr_t)gbase;
    const unsigned int lds = (unsigned int)(uintptr_t)((wv == 0) ? &Qs[0] : &Ks[0]);
    // D# group0: count=1, lds_addr, 57-bit global addr, type=2 (bits 127:126)
    v4u g0 = { 1u, lds, (unsigned int)(ga & 0xffffffffu),
               (unsigned int)((ga >> 32) & 0x1ffffffu) | (2u << 30) };
    // D# group1: data_size=2B; tensor_dim0=64, tensor_dim1=128;
    //            tile_dim0=64, tile_dim1=128; tensor_dim0_stride=512
    v8i g1 = { (int)(1u << 16),    // [17:16] data_size=1 (2 bytes)
               (int)(64u << 16),   // [79:48] tensor_dim0 = 64 (low 16)
               (int)(128u << 16),  // [111:80] tensor_dim1 = 128 (low 16)
               (int)(64u << 16),   // [127:112] tile_dim0 = 64
               (int)128,           // [143:128] tile_dim1 = 128; tile_dim2 = 0
               (int)512,           // [207:160] tensor_dim0_stride = 512
               0, 0 };
    v4i gz4 = { 0, 0, 0, 0 };
    v8i gz8 = { 0, 0, 0, 0, 0, 0, 0, 0 };
    __builtin_amdgcn_tensor_load_to_lds(g0, g1, gz4, gz4, gz8, 0);
    __builtin_amdgcn_s_wait_tensorcnt(0);
  }
  __syncthreads();
#else
  {  // manual: 2 threads/row, 32 halfs each for both Q and K tiles
    const int r = t >> 1, c = (t & 1) * 32;
    const h16* gq = q + (size_t)(b * SEQ + qBase + r) * DIM + h * MD + c;
    const h16* gk = k + (size_t)(b * SEQ + kBase + r) * DIM + h * MD + c;
#pragma unroll
    for (int i = 0; i < 4; ++i) {
      *(h8*)&Qs[r * 64 + c + i * 8] = *(const h8*)(gq + i * 8);
      *(h8*)&Ks[r * 64 + c + i * 8] = *(const h8*)(gk + i * 8);
    }
  }
  __syncthreads();
#endif

  const int wm = (wv & 3) * 32;    // 4 waves over 128 q-rows
  const int wn = (wv >> 2) * 64;   // 2 waves over 128 k-cols
  const int hs = lane >> 4, lr = lane & 15;
  v8f acc[2][4] = {};
#pragma unroll
  for (int dd = 0; dd < 64; dd += 32) {
    v16h af[2], bf[4];
#pragma unroll
    for (int mi = 0; mi < 2; ++mi) af[mi] = load_a_frag(Qs, wm + mi * 16 + lr, 64, dd, hs);
#pragma unroll
    for (int ni = 0; ni < 4; ++ni) bf[ni] = load_b_frag(Ks, wn + ni * 16 + lr, 64, dd, hs);
#pragma unroll
    for (int mi = 0; mi < 2; ++mi)
#pragma unroll
      for (int ni = 0; ni < 4; ++ni) acc[mi][ni] = WMMA_F16(af[mi], bf[ni], acc[mi][ni]);
  }
  const float scale = 0.125f;  // 1/sqrt(64)
#pragma unroll
  for (int mi = 0; mi < 2; ++mi)
#pragma unroll
    for (int ni = 0; ni < 4; ++ni) {
      const int col = kBase + wn + ni * 16 + lr;
      const int rB = qBase + wm + mi * 16 + hs * 8;
#pragma unroll
      for (int r = 0; r < 8; ++r)
        corr[((size_t)(b * SEQ + rB + r) * HEADS + h) * SEQ + col] = acc[mi][ni][r] * scale;
    }
}

// ---------------------------------------------------------------------------
// 5) Softmax in place over each 1024-wide row of attn (65536 rows).
// ---------------------------------------------------------------------------
__global__ __launch_bounds__(256) void softmax_kernel(float* __restrict__ attn) {
  __shared__ float red[256];
  float* row = attn + (size_t)blockIdx.x * SEQ;
  const int t = threadIdx.x;
  float v[4];
  float m = -1e30f;
#pragma unroll
  for (int i = 0; i < 4; ++i) { v[i] = row[t + i * 256]; m = fmaxf(m, v[i]); }
  red[t] = m;
  __syncthreads();
#pragma unroll
  for (int off = 128; off > 0; off >>= 1) {
    if (t < off) red[t] = fmaxf(red[t], red[t + off]);
    __syncthreads();
  }
  m = red[0];
  __syncthreads();
  float s = 0.f;
#pragma unroll
  for (int i = 0; i < 4; ++i) { v[i] = __expf(v[i] - m); s += v[i]; }
  red[t] = s;
  __syncthreads();
#pragma unroll
  for (int off = 128; off > 0; off >>= 1) {
    if (t < off) red[t] += red[t + off];
    __syncthreads();
  }
  const float inv = 1.0f / red[0];
#pragma unroll
  for (int i = 0; i < 4; ++i) row[t + i * 256] = v[i] * inv;
}

// ---------------------------------------------------------------------------
// 6) attn(f32 -> f16 while staging) @ V -> logits f16 [8192x512].
//    V arrives pre-transposed (vT), so B staging is contiguous b128 moves.
// ---------------------------------------------------------------------------
__global__ __launch_bounds__(256) void av_kernel(const float* __restrict__ attn,
                                                 const h16* __restrict__ vT,
                                                 h16* __restrict__ logits) {
  __shared__ __attribute__((aligned(16))) h16 As[128 * 32];
  __shared__ __attribute__((aligned(16))) h16 Vt[64 * 32];
  const int mBase = blockIdx.x * 128;
  const int bh = blockIdx.y;
  const int b = bh >> 3, h = bh & 7;
  const int t = threadIdx.x;
  const int lane = t & 31, wv = t >> 5;
  const int wm = (wv & 3) * 32, wn = (wv >> 2) * 32;
  const int hs = lane >> 4, lr = lane & 15;

  v8f acc[2][2] = {};
  for (int jj = 0; jj < SEQ; jj += 32) {
    {  // attn tile 128x32 f32 -> f16, packed h8 stores
      const int r = t >> 1, c = (t & 1) * 16;
      const f4* g = (const f4*)(attn + ((size_t)(b * SEQ + mBase + r) * HEADS + h) * SEQ + jj + c);
      f4 x0 = g[0], x1 = g[1], x2 = g[2], x3 = g[3];
      h8 p0 = { (h16)x0[0], (h16)x0[1], (h16)x0[2], (h16)x0[3],
                (h16)x1[0], (h16)x1[1], (h16)x1[2], (h16)x1[3] };
      h8 p1 = { (h16)x2[0], (h16)x2[1], (h16)x2[2], (h16)x2[3],
                (h16)x3[0], (h16)x3[1], (h16)x3[2], (h16)x3[3] };
      *(h8*)&As[r * 32 + c] = p0;
      *(h8*)&As[r * 32 + c + 8] = p1;
      if (jj + 32 < SEQ) __builtin_prefetch(g + 8, 0, 3);
    }
    {  // Vt tile 64(d) x 32(j): contiguous from pre-transposed V
      const int d = t >> 2, jo = (t & 3) * 8;
      *(h8*)&Vt[d * 32 + jo] =
          *(const h8*)(vT + ((size_t)(b * 512 + h * MD + d)) * SEQ + jj + jo);
    }
    __syncthreads();
    v16h af[2], bf[2];
#pragma unroll
    for (int mi = 0; mi < 2; ++mi) af[mi] = load_a_frag(As, wm + mi * 16 + lr, 32, 0, hs);
#pragma unroll
    for (int ni = 0; ni < 2; ++ni) bf[ni] = load_b_frag(Vt, wn + ni * 16 + lr, 32, 0, hs);
#pragma unroll
    for (int mi = 0; mi < 2; ++mi)
#pragma unroll
      for (int ni = 0; ni < 2; ++ni) acc[mi][ni] = WMMA_F16(af[mi], bf[ni], acc[mi][ni]);
    __syncthreads();
  }
#pragma unroll
  for (int mi = 0; mi < 2; ++mi)
#pragma unroll
    for (int ni = 0; ni < 2; ++ni) {
      const int col = h * MD + wn + ni * 16 + lr;
      const int rB = b * SEQ + mBase + wm + mi * 16 + hs * 8;
#pragma unroll
      for (int r = 0; r < 8; ++r)
        logits[(size_t)(rB + r) * DIM + col] = (h16)acc[mi][ni][r];
    }
}

// ---------------------------------------------------------------------------
// 7) Out projection: logits f16 @ Wo f16 + xn -> f32 out (Wo pre-transposed).
// ---------------------------------------------------------------------------
__global__ __launch_bounds__(256) void out_kernel(const h16* __restrict__ logits,
                                                  const h16* __restrict__ woT,
                                                  const float* __restrict__ xn,
                                                  float* __restrict__ out) {
  __shared__ __attribute__((aligned(16))) h16 As[128 * 32];
  __shared__ __attribute__((aligned(16))) h16 Bt[64 * 32];
  const int mBase = blockIdx.x * 128;
  const int nBase = blockIdx.y * 64;
  const int t = threadIdx.x;
  const int lane = t & 31, wv = t >> 5;
  const int wm = (wv & 3) * 32, wn = (wv >> 2) * 32;
  const int hs = lane >> 4, lr = lane & 15;

  v8f acc[2][2] = {};
  for (int kk = 0; kk < DIM; kk += 32) {
    {
      const int r = t >> 1, c = (t & 1) * 16;
      const h16* g = logits + (size_t)(mBase + r) * DIM + kk + c;
      *(h8*)&As[r * 32 + c] = *(const h8*)(g);
      *(h8*)&As[r * 32 + c + 8] = *(const h8*)(g + 8);
      if (kk + 32 < DIM) __builtin_prefetch(g + 32, 0, 3);
    }
    {
      const int n = t >> 2, ko = (t & 3) * 8;
      *(h8*)&Bt[n * 32 + ko] = *(const h8*)(woT + (size_t)(nBase + n) * DIM + kk + ko);
    }
    __syncthreads();
    v16h af[2], bf[2];
#pragma unroll
    for (int mi = 0; mi < 2; ++mi) af[mi] = load_a_frag(As, wm + mi * 16 + lr, 32, 0, hs);
#pragma unroll
    for (int ni = 0; ni < 2; ++ni) bf[ni] = load_b_frag(Bt, wn + ni * 16 + lr, 32, 0, hs);
#pragma unroll
    for (int mi = 0; mi < 2; ++mi)
#pragma unroll
      for (int ni = 0; ni < 2; ++ni) acc[mi][ni] = WMMA_F16(af[mi], bf[ni], acc[mi][ni]);
    __syncthreads();
  }
#pragma unroll
  for (int mi = 0; mi < 2; ++mi)
#pragma unroll
    for (int ni = 0; ni < 2; ++ni) {
      const int col = nBase + wn + ni * 16 + lr;
      const int rB = mBase + wm + mi * 16 + hs * 8;
#pragma unroll
      for (int r = 0; r < 8; ++r) {
        const size_t idx = (size_t)(rB + r) * DIM + col;
        out[idx] = acc[mi][ni][r] + xn[idx];  // residual uses normalized x
      }
    }
}

// ---------------------------------------------------------------------------
extern "C" void kernel_launch(void* const* d_in, const int* in_sizes, int n_in,
                              void* d_out, int out_size, void* d_ws, size_t ws_size,
                              hipStream_t stream) {
  const float* x     = (const float*)d_in[0];
  const float* Wq    = (const float*)d_in[1];
  const float* Wk    = (const float*)d_in[2];
  const float* Wv    = (const float*)d_in[3];
  const float* Wo    = (const float*)d_in[4];
  const float* gamma = (const float*)d_in[5];
  const float* beta  = (const float*)d_in[6];

  float* out  = (float*)d_out;
  float* attn = out + (size_t)ROWS * DIM;  // tuple: (out[8192*512], attn[65536*1024])

  char* ws = (char*)d_ws;
  float* xn  = (float*)(ws + OFF_XN);
  h16* xnh   = (h16*)(ws + OFF_XNH);
  h16* qkv   = (h16*)(ws + OFF_QKV);
  h16* logit = (h16*)(ws + OFF_LOG);
  h16* wh    = (h16*)(ws + OFF_WH);

  ln_kernel<<<ROWS, 256, 0, stream>>>(x, gamma, beta, xn, xnh);
  wcast_kernel<<<4096, 256, 0, stream>>>(Wq, Wk, Wv, Wo, wh);
  gemm_qkv_kernel<<<dim3(ROWS / 128, DIM / 64, 3), 256, 0, stream>>>(xnh, wh, qkv);
  scores_kernel<<<dim3(SEQ / 128, SEQ / 128, BATCH * HEADS), 256, 0, stream>>>(
      qkv, qkv + (size_t)ROWS * DIM, attn);
  softmax_kernel<<<BATCH * SEQ * HEADS, 256, 0, stream>>>(attn);
  av_kernel<<<dim3(SEQ / 128, BATCH * HEADS), 256, 0, stream>>>(
      attn, qkv + 2 * (size_t)ROWS * DIM, logit);
  out_kernel<<<dim3(ROWS / 128, DIM / 64), 256, 0, stream>>>(logit, wh + 3 * (size_t)DIM * DIM, xn, out);
}
